// NRField_4320737099877
// MI455X (gfx1250) — matrix operand order repaired and checked by reading, not compiled, not run on hardware
//
#include <hip/hip_runtime.h>
#include <hip/hip_bf16.h>

typedef __attribute__((ext_vector_type(16))) _Float16 v16h;
typedef __attribute__((ext_vector_type(8)))  _Float16 v8h;
typedef __attribute__((ext_vector_type(8)))  float    v8f;

#define LDA 256              // activation LDS row stride (halfs)
#define TMASK ((1u << 22) - 1u)

// LDS layout (bytes, dynamic shared assumed to start at LDS addr 0):
//   act   @ 0       : 128*256 halfs = 65536 B
//   wbuf0 @ 65536   : 65536 halfs  = 131072 B
//   wbuf1 @ 196608  : 65536 halfs  = 131072 B
//   total 327680 B = 320 KB (exactly the WGP LDS)
#define ACT_OFF_B   0u
#define WBUF0_OFF_B 65536u
#define WBUF1_OFF_B 196608u
#define SMEM_BYTES  327680u

// ---------------------------------------------------------------------------
// Weight prep: f32 -> f16, pad W0 to [256][64], W4 to [16][256].
// ws layout (halfs): W0p @0 (16384), W1 @16384, W2 @81920, W3 @147456,
//                    W4p @212992 (4096). total 217088 halfs.
// ---------------------------------------------------------------------------
__global__ void prep_weights(const float* __restrict__ W0, const float* __restrict__ W1,
                             const float* __restrict__ W2, const float* __restrict__ W3,
                             const float* __restrict__ W4, _Float16* __restrict__ wh) {
  int i = blockIdx.x * 256 + threadIdx.x;
  float v;
  if (i < 16384) {                       // W0 pad: [256][64] from [256][44]
    int row = i >> 6, col = i & 63;
    v = (col < 44) ? W0[row * 44 + col] : 0.0f;
  } else if (i < 81920) {
    v = W1[i - 16384];
  } else if (i < 147456) {
    v = W2[i - 81920];
  } else if (i < 212992) {
    v = W3[i - 147456];
  } else if (i < 217088) {               // W4 pad: [16][256] from [3][256]
    int j = i - 212992;
    int row = j >> 8, col = j & 255;
    v = (row < 3) ? W4[row * 256 + col] : 0.0f;
  } else {
    return;
  }
  wh[i] = (_Float16)v;
}

// ---------------------------------------------------------------------------
// CDNA5 async global->LDS copy (tracked by ASYNCcnt), inline asm so it is
// portable across ROCm-7.2 / amdgpu-toolchain builtin arity differences.
// ---------------------------------------------------------------------------
__device__ __forceinline__ void async_copy_b128(unsigned lds_byte_off,
                                                const void* gptr) {
  asm volatile("global_load_async_to_lds_b128 %0, %1, off"
               :: "v"(lds_byte_off), "v"((unsigned long long)gptr)
               : "memory");
}

template <int N>
__device__ __forceinline__ void wait_asynccnt() {
  asm volatile("s_wait_asynccnt %0" :: "n"(N) : "memory");
}

// Stage CHUNKS*16 bytes cooperatively (256 threads, b128 per thread per iter).
template <int CHUNKS_PER_THREAD>
__device__ __forceinline__ void stage_weights(unsigned lds_base_b,
                                              const _Float16* __restrict__ src,
                                              int tid) {
#pragma unroll
  for (int j = 0; j < CHUNKS_PER_THREAD; ++j) {
    const int c = tid + 256 * j;
    async_copy_b128(lds_base_b + (unsigned)c * 16u,
                    (const void*)((const char*)src + (size_t)c * 16u));
  }
}

// ---------------------------------------------------------------------------
// WMMA fragment loaders (contiguous k = 16*hi + s, same order for A and B)
// ---------------------------------------------------------------------------
// A fragment (16x32 f16, MxK): lanes 0-15 -> M=lane; lanes 16-31 same M, K+16.
// Per lane one contiguous 32B run: [kb+16*hi, +16).
__device__ __forceinline__ v16h load_afrag(const _Float16* __restrict__ rowp,
                                           int kb, int hi) {
  return *(const v16h*)(rowp + kb + hi * 16);
}

// B fragment (32x16 f16, KxN): lanes 0-15 -> N=lane; lanes 16-31 same N, K+16;
// linear K packing -> 16 contiguous halfs (32 B) per lane.
template <int KDIM>
__device__ __forceinline__ v16h load_bfrag(const _Float16* __restrict__ W,
                                           int nn, int kb, int hi) {
  return *(const v16h*)(W + nn * KDIM + kb + hi * 16);
}

// GEMM over this wave's 16-row strip: acc[n] += A(16xK) * W^T tile n.
// W now points into LDS (staged weights) -> B-frags are ds_load_b128 pairs.
template <int KDIM, int NTILES>
__device__ __forceinline__ void wmma_rows(const _Float16* __restrict__ actIn,
                                          const _Float16* __restrict__ W,
                                          v8f acc[NTILES], int mrow, int lane) {
  const int hi = lane >> 4;
  const int nlo = lane & 15;
#pragma unroll
  for (int n = 0; n < NTILES; ++n)
#pragma unroll
    for (int e = 0; e < 8; ++e) acc[n][e] = 0.0f;

  const _Float16* rowp = actIn + mrow * LDA;
  for (int kb = 0; kb < KDIM; kb += 32) {
    v16h a = load_afrag(rowp, kb, hi);
#pragma unroll
    for (int n = 0; n < NTILES; ++n) {
      v16h b = load_bfrag<KDIM>(W, n * 16 + nlo, kb, hi);
      acc[n] = __builtin_amdgcn_wmma_f32_16x16x32_f16(
          false, a, false, b, (short)0, acc[n], false, false);
    }
  }
}

// bias + ReLU, convert to f16, scatter back to activation LDS (own rows only).
// C/D layout: VGPR r -> M = r + 8*hi, lane -> N = (lane&15) within tile.
__device__ __forceinline__ void store_relu(const v8f acc[16], _Float16* __restrict__ act,
                                           const float* __restrict__ bias,
                                           int m16, int lane) {
  const int hi = lane >> 4, nlo = lane & 15;
#pragma unroll
  for (int n = 0; n < 16; ++n) {
    const int nn = n * 16 + nlo;
    const float bv = bias[nn];
#pragma unroll
    for (int r = 0; r < 8; ++r) {
      float v = acc[n][r] + bv;
      v = v > 0.0f ? v : 0.0f;
      act[(m16 + r + hi * 8) * LDA + nn] = (_Float16)v;
    }
  }
}

// ---------------------------------------------------------------------------
// Fused kernel: hash-grid encode -> 5-layer MLP (WMMA f16, LDS-resident
// weights double-buffered via async DMA) -> |.| output.
// 256 threads = 8 waves, 128 points per block; rows are wave-private.
// ---------------------------------------------------------------------------
__global__ __launch_bounds__(256)
void nrfield_fused(const float* __restrict__ x, const float* __restrict__ wvi,
                   const float* __restrict__ nrm, const float* __restrict__ fd,
                   const float* __restrict__ table,
                   const _Float16* __restrict__ wh,
                   const float* __restrict__ b0, const float* __restrict__ b1,
                   const float* __restrict__ b2, const float* __restrict__ b3,
                   const float* __restrict__ b4,
                   float* __restrict__ out) {
  extern __shared__ _Float16 smem[];
  _Float16* act   = smem;                    // 128*256 halfs
  _Float16* wbuf0 = smem + 32768;            // 65536 halfs
  _Float16* wbuf1 = smem + 98304;            // 65536 halfs

  const int tid = threadIdx.x;
  const int lane = tid & 31;
  const int wave = tid >> 5;

  const _Float16* W0h = wh;                  // [256][64]  = 2048 x 16B
  const _Float16* W1h = wh + 16384;          // [256][256] = 8192 x 16B
  const _Float16* W2h = wh + 81920;
  const _Float16* W3h = wh + 147456;
  const _Float16* W4h = wh + 212992;         // [16][256]  = 512 x 16B

  // Kick off W0 -> wbuf0 (8 per thread) and W1 -> wbuf1 (32 per thread);
  // the hash-gather encode below hides the DMA latency.
  stage_weights<8>(WBUF0_OFF_B, W0h, tid);
  stage_weights<32>(WBUF1_OFF_B, W1h, tid);

  // ---------------- encode: 2 threads per point, 4 levels each -------------
  {
    const int p = tid >> 1;              // 0..127 (row p belongs to wave p>>4)
    const int half = tid & 1;
    const int gp = blockIdx.x * 128 + p;
    const float X0 = x[gp * 3 + 0], X1 = x[gp * 3 + 1], X2 = x[gp * 3 + 2];
    _Float16* row = &act[p * LDA];
    if (half == 0) {
      row[0] = (_Float16)X0; row[1] = (_Float16)X1; row[2] = (_Float16)X2;
      row[3] = (_Float16)wvi[gp * 3 + 0]; row[4] = (_Float16)wvi[gp * 3 + 1];
      row[5] = (_Float16)wvi[gp * 3 + 2];
      row[6] = (_Float16)nrm[gp * 3 + 0]; row[7] = (_Float16)nrm[gp * 3 + 1];
      row[8] = (_Float16)nrm[gp * 3 + 2];
      row[9] = (_Float16)fd[gp * 3 + 0]; row[10] = (_Float16)fd[gp * 3 + 1];
      row[11] = (_Float16)fd[gp * 3 + 2];
    } else {
#pragma unroll
      for (int c = 44; c < 64; ++c) row[c] = (_Float16)0.0f;  // K-pad
    }
#pragma unroll
    for (int li = 0; li < 4; ++li) {
      const int l = half * 4 + li;
      const float res = (float)(16 << l);
      const float p0 = X0 * res, p1 = X1 * res, p2 = X2 * res;
      const float f0 = floorf(p0), f1 = floorf(p1), f2 = floorf(p2);
      const float w0 = p0 - f0, w1 = p1 - f1, w2 = p2 - f2;
      const unsigned ix = (unsigned)(int)f0, iy = (unsigned)(int)f1,
                     iz = (unsigned)(int)f2;
      const unsigned hx0 = ix, hx1 = ix + 1u;
      const unsigned hy0 = iy * 2654435761u, hy1 = (iy + 1u) * 2654435761u;
      const unsigned hz0 = iz * 805459861u,  hz1 = (iz + 1u) * 805459861u;
      const float4* tl = (const float4*)table + ((size_t)l << 22);
      float a0 = 0.f, a1 = 0.f, a2 = 0.f, a3 = 0.f;
#pragma unroll
      for (int c = 0; c < 8; ++c) {
        const unsigned h =
            (((c & 4) ? hx1 : hx0) ^ ((c & 2) ? hy1 : hy0) ^
             ((c & 1) ? hz1 : hz0)) & TMASK;
        const float4 g = tl[h];                       // global_load_b128 gather
        const float cw = ((c & 4) ? w0 : 1.f - w0) *
                         ((c & 2) ? w1 : 1.f - w1) *
                         ((c & 1) ? w2 : 1.f - w2);
        a0 = fmaf(g.x, cw, a0); a1 = fmaf(g.y, cw, a1);
        a2 = fmaf(g.z, cw, a2); a3 = fmaf(g.w, cw, a3);
      }
      row[12 + l * 4 + 0] = (_Float16)a0;
      row[12 + l * 4 + 1] = (_Float16)a1;
      row[12 + l * 4 + 2] = (_Float16)a2;
      row[12 + l * 4 + 3] = (_Float16)a3;
    }
  }

  // ---------------- MLP ----------------------------------------------------
  const int hi = lane >> 4;
  const int nlo = lane & 15;
  const int m16 = wave * 16;             // this wave's 16-row strip
  const int mrow = m16 + nlo;            // A-matrix row for this lane

  v8f acc[16];

  // W0 landed when <=32 remain (in-order completion: only W1's 32 can remain).
  wait_asynccnt<32>();
  __syncthreads();

  // layer 0: K=64 (44 padded), N=256, weights in wbuf0
  wmma_rows<64, 16>(act, wbuf0, acc, mrow, lane);
  __syncthreads();                       // everyone done reading wbuf0
  store_relu(acc, act, b0, m16, lane);   // own rows: no extra barrier needed
  stage_weights<32>(WBUF0_OFF_B, W2h, tid);  // refill wbuf0 with W2
  wait_asynccnt<32>();                   // W1 done (only W2's 32 remain)
  __syncthreads();

  // layer 1: wbuf1 (W1)
  wmma_rows<256, 16>(act, wbuf1, acc, mrow, lane);
  __syncthreads();
  store_relu(acc, act, b1, m16, lane);
  stage_weights<32>(WBUF1_OFF_B, W3h, tid);  // refill wbuf1 with W3
  wait_asynccnt<32>();                   // W2 done
  __syncthreads();

  // layer 2: wbuf0 (W2)
  wmma_rows<256, 16>(act, wbuf0, acc, mrow, lane);
  __syncthreads();
  store_relu(acc, act, b2, m16, lane);
  stage_weights<2>(WBUF0_OFF_B, W4h, tid);   // refill wbuf0 with W4 (8 KB)
  wait_asynccnt<2>();                    // W3 done
  __syncthreads();

  // layer 3: wbuf1 (W3)
  wmma_rows<256, 16>(act, wbuf1, acc, mrow, lane);
  __syncthreads();
  store_relu(acc, act, b3, m16, lane);
  wait_asynccnt<0>();                    // W4 done
  __syncthreads();

  // layer 4: K=256, N=16 (3 real cols), out = |h @ W4^T + b4|
  v8f o[1];
  wmma_rows<256, 1>(act, wbuf0, o, mrow, lane);
  if (nlo < 3) {
    const float bv = b4[nlo];
#pragma unroll
    for (int r = 0; r < 8; ++r) {
      const int m = m16 + r + hi * 8;
      out[((size_t)blockIdx.x * 128 + m) * 3 + nlo] = fabsf(o[0][r] + bv);
    }
  }
}

// ---------------------------------------------------------------------------
extern "C" void kernel_launch(void* const* d_in, const int* in_sizes, int n_in,
                              void* d_out, int out_size, void* d_ws, size_t ws_size,
                              hipStream_t stream) {
  const float* x   = (const float*)d_in[0];
  const float* wi  = (const float*)d_in[1];
  const float* nrm = (const float*)d_in[2];
  const float* fd  = (const float*)d_in[3];
  const float* tbl = (const float*)d_in[4];
  const float* W0 = (const float*)d_in[5];  const float* b0 = (const float*)d_in[6];
  const float* W1 = (const float*)d_in[7];  const float* b1 = (const float*)d_in[8];
  const float* W2 = (const float*)d_in[9];  const float* b2 = (const float*)d_in[10];
  const float* W3 = (const float*)d_in[11]; const float* b3 = (const float*)d_in[12];
  const float* W4 = (const float*)d_in[13]; const float* b4 = (const float*)d_in[14];
  float* out = (float*)d_out;
  _Float16* wh = (_Float16*)d_ws;        // 217088 halfs = 434 KB scratch

  const int N = in_sizes[0] / 3;         // 524288

  prep_weights<<<848, 256, 0, stream>>>(W0, W1, W2, W3, W4, wh);
  nrfield_fused<<<N / 128, 256, SMEM_BYTES, stream>>>(x, wi, nrm, fd, tbl, wh,
                                                      b0, b1, b2, b3, b4, out);
}
